// refine_pnplus_87101936763050
// MI455X (gfx1250) — matrix operand order repaired and checked by reading, not compiled, not run on hardware
//
#include <hip/hip_runtime.h>

typedef __attribute__((ext_vector_type(16))) _Float16 v16h;
typedef __attribute__((ext_vector_type(8)))  _Float16 v8h;
typedef __attribute__((ext_vector_type(8)))  float    v8f;

#define GF_BIAS 1
#define GF_BN   2
#define GF_RELU 4
#define GF_RES  8
#define GF_TANH 16

// ---------------------------------------------------------------------------
// Tiled GEMM on v_wmma_f32_16x16x32_f16 (wave32, 4 waves/block).
//   Y[b][o][m] = epilogue( sum_c W[o][c] * X[b][c][m] )
// Block tile: 16 (O) x 64 (M) x 64 (K-step); each wave owns one 16x16 WMMA
// accumulator and issues two WMMAs per staging round.
// Staging is branch-free: clamped addresses + value predication, loads
// gathered to registers first, f32->f16 packed with v_cvt_pk, 16B LDS stores.
// ldX/ldY/ldR allow strided sub-views (chunked attention MLP).
// ---------------------------------------------------------------------------
typedef decltype(__builtin_amdgcn_cvt_pkrtz(0.f, 0.f)) pkh2;
union PK8 { v8h v; pkh2 p[4]; };

__global__ __launch_bounds__(128) void gemm_wmma_kernel(
    const float* __restrict__ W, const float* __restrict__ X,
    const float* __restrict__ bias, const float* __restrict__ gamma,
    const float* __restrict__ beta, const float* __restrict__ Res,
    float* __restrict__ Y, int O, int C, int M,
    long ldX, long ldY, long ldR, int flags)
{
  __shared__ __align__(16) _Float16 sW[16 * 64];   // [o][k]
  __shared__ __align__(16) _Float16 sX[64 * 64];   // [m][k] (transposed)
  const int tid  = threadIdx.x;
  const int lane = tid & 31;
  const int wave = tid >> 5;
  const int b  = blockIdx.z;
  const int O0 = blockIdx.y * 16;
  const int M0 = blockIdx.x * 64;
  const float* Xb = X + (long)b * C * ldX;

  // staging coordinates (fixed per thread)
  const int  wo   = tid >> 3;                       // W row 0..15
  const int  wk   = (tid & 7) * 8;                  // W k-base in tile
  const bool wok  = (O0 + wo) < O;
  const long wrow = (long)(wok ? (O0 + wo) : (O - 1)) * C;

  const int  xm   = tid & 63;                       // X column (m) 0..63
  const int  xk   = (tid >> 6) * 32;                // X k-base in tile (0/32)
  const bool mok  = (M0 + xm) < M;
  const long mc   = (long)(mok ? (M0 + xm) : (M - 1));

  v8f acc = {};
  for (int k0 = 0; k0 < C; k0 += 64) {
    if (k0 + 64 < C) {   // uniform branch: prefetch next K tile while we compute
      __builtin_prefetch(&Xb[(long)(k0 + 64) * ldX + M0], 0, 1);
      __builtin_prefetch(&W[wrow + k0 + 64], 0, 1);
    }
    // ---- gather W slice: one row, 8 consecutive k (always-valid addresses)
    float wv[8];
    #pragma unroll
    for (int u = 0; u < 8; ++u) {
      int kc = k0 + wk + u;
      wv[u] = W[wrow + (kc < C ? kc : 0)];
    }
    // ---- gather X slice: one column, 32 consecutive k
    float xv[32];
    #pragma unroll
    for (int u = 0; u < 32; ++u) {
      int kc = k0 + xk + u;
      xv[u] = Xb[(long)(kc < C ? kc : 0) * ldX + mc];
    }
    __syncthreads();   // previous round's fragment reads complete
    {
      PK8 uw;
      #pragma unroll
      for (int u = 0; u < 4; ++u) {
        float a0 = (wok && (k0 + wk + 2 * u)     < C) ? wv[2 * u]     : 0.f;
        float a1 = (wok && (k0 + wk + 2 * u + 1) < C) ? wv[2 * u + 1] : 0.f;
        uw.p[u] = __builtin_amdgcn_cvt_pkrtz(a0, a1);
      }
      *(v8h*)&sW[wo * 64 + wk] = uw.v;               // one b128 store
    }
    #pragma unroll
    for (int q = 0; q < 4; ++q) {
      PK8 ux;
      #pragma unroll
      for (int u = 0; u < 4; ++u) {
        int kk = q * 8 + 2 * u;
        float a0 = (mok && (k0 + xk + kk)     < C) ? xv[kk]     : 0.f;
        float a1 = (mok && (k0 + xk + kk + 1) < C) ? xv[kk + 1] : 0.f;
        ux.p[u] = __builtin_amdgcn_cvt_pkrtz(a0, a1);
      }
      *(v8h*)&sX[xm * 64 + xk + q * 8] = ux.v;       // four b128 stores total
    }
    __syncthreads();
    // ---- two 16x16x32 WMMA steps over the 64-wide K tile ----
    const int r  = lane & 15;           // A row
    const int kh = lane >> 4;           // A K-half select
    const int ml = wave * 16 + (lane & 15);   // B column in block M tile
    const int kb = (lane >> 4) * 8;           // B K base
    #pragma unroll
    for (int hseg = 0; hseg < 2; ++hseg) {
      const int ks = hseg * 32;
      v8h a0 = *(const v8h*)&sW[r * 64 + ks + kh * 8];
      v8h a1 = *(const v8h*)&sW[r * 64 + ks + 16 + kh * 8];
      v16h af = __builtin_shufflevector(a0, a1, 0,1,2,3,4,5,6,7,8,9,10,11,12,13,14,15);
      v8h b0 = *(const v8h*)&sX[ml * 64 + ks + kb];
      v8h b1 = *(const v8h*)&sX[ml * 64 + ks + kb + 16];
      v16h bf = __builtin_shufflevector(b0, b1, 0,1,2,3,4,5,6,7,8,9,10,11,12,13,14,15);
      acc = __builtin_amdgcn_wmma_f32_16x16x32_f16(false, af, false, bf, (short)0, acc,
                                                   false, false);
    }
  }
  const int cc = lane & 15;
  const int m  = M0 + wave * 16 + cc;
  if (m < M) {
    #pragma unroll
    for (int i = 0; i < 8; ++i) {
      int o = O0 + i + (lane >> 4) * 8;   // C/D layout: VGPR i -> M=i / i+8
      if (o < O) {
        float v = acc[i];
        if (flags & GF_BIAS) v += bias[o];
        if (flags & GF_BN)   v = v * gamma[o] + beta[o];
        if (flags & GF_RELU) v = v > 0.f ? v : 0.f;
        if (flags & GF_RES)  v += Res[(long)b * O * ldR + (long)o * ldR + m];
        if (flags & GF_TANH) v = tanhf(v);
        Y[(long)b * O * ldY + (long)o * ldY + m] = v;
      }
    }
  }
}

// ---------------------------------------------------------------------------
// Farthest point sampling: one block per batch, dist array resident in LDS.
// Matches the reference scan (emit current 'far', then update / argmax).
// ---------------------------------------------------------------------------
__global__ void k_fps(const float* __restrict__ xyz, int N, int npoint,
                      int* __restrict__ out)
{
  __shared__ float dist[2048];
  __shared__ float rv[256];
  __shared__ int   ri[256];
  __shared__ int   far_s;
  const int b = blockIdx.x, tid = threadIdx.x;
  const float* px = xyz + (long)b * 3 * N;
  for (int i = tid; i < N; i += 256) dist[i] = 1e10f;
  if (tid == 0) far_s = 0;
  __syncthreads();
  for (int it = 0; it < npoint; ++it) {
    int far = far_s;
    if (tid == 0) out[b * npoint + it] = far;
    float cx = px[far], cy = px[N + far], cz = px[2 * N + far];
    float best = -1.f; int bi = 0;
    for (int i = tid; i < N; i += 256) {
      float dx = px[i] - cx, dy = px[N + i] - cy, dz = px[2 * N + i] - cz;
      float d = dx * dx + dy * dy + dz * dz;
      float dm = dist[i]; dm = d < dm ? d : dm; dist[i] = dm;
      if (dm > best) { best = dm; bi = i; }
    }
    rv[tid] = best; ri[tid] = bi;
    __syncthreads();
    for (int s = 128; s > 0; s >>= 1) {
      if (tid < s) {
        if (rv[tid + s] > rv[tid] ||
            (rv[tid + s] == rv[tid] && ri[tid + s] < ri[tid])) {
          rv[tid] = rv[tid + s]; ri[tid] = ri[tid + s];
        }
      }
      __syncthreads();
    }
    if (tid == 0) far_s = ri[0];
    __syncthreads();
  }
}

// gather sampled coordinates: out[B,3,J]
__global__ void k_gather_xyz(const float* __restrict__ xyz, const int* __restrict__ idx,
                             int N, int J, float* __restrict__ out, long total)
{
  long t = blockIdx.x * 256L + threadIdx.x; if (t >= total) return;
  int j = (int)(t % J); int c = (int)((t / J) % 3); int b = (int)(t / (3L * J));
  out[t] = xyz[((long)b * 3 + c) * N + idx[(long)b * J + j]];
}

// ball query: first K in-radius indices in ascending order, padded with first
__global__ void k_ball_query(const float* __restrict__ xyz, const float* __restrict__ nx,
                             int N, int J, int K, float r2, int* __restrict__ idx,
                             long total)
{
  long t = blockIdx.x * 256L + threadIdx.x; if (t >= total) return;
  int b = (int)(t / J), j = (int)(t % J);
  const float* px = xyz + (long)b * 3 * N;
  float cx = nx[((long)b * 3 + 0) * J + j];
  float cy = nx[((long)b * 3 + 1) * J + j];
  float cz = nx[((long)b * 3 + 2) * J + j];
  int* o = idx + ((long)b * J + j) * K;
  int cnt = 0, first = 0;
  for (int i = 0; i < N && cnt < K; ++i) {
    float dx = px[i] - cx, dy = px[N + i] - cy, dz = px[2 * N + i] - cz;
    float d = dx * dx + dy * dy + dz * dz;
    if (d <= r2) { if (cnt == 0) first = i; o[cnt++] = i; }
  }
  for (; cnt < K; ++cnt) o[cnt] = first;
}

// build grouped input: out[B, 3+D, K, J] = concat(rel_xyz, grouped points)
__global__ void k_group_sa(const float* __restrict__ xyz, const float* __restrict__ nx,
                           const float* __restrict__ pts, const int* __restrict__ idx,
                           int N, int J, int K, int D, float* __restrict__ out, long total)
{
  long t = blockIdx.x * 256L + threadIdx.x; if (t >= total) return;
  int Cin = 3 + D;
  int j = (int)(t % J); long r = t / J;
  int kk = (int)(r % K); r /= K;
  int c = (int)(r % Cin); int b = (int)(r / Cin);
  int id = idx[((long)b * J + j) * K + kk];
  float v;
  if (c < 3) v = xyz[((long)b * 3 + c) * N + id] - nx[((long)b * 3 + c) * J + j];
  else       v = pts[((long)b * D + (c - 3)) * N + id];
  out[t] = v;
}

// max over K: in [B,C,K,J] -> out [B,C,J]
__global__ void k_maxpool(const float* __restrict__ x, int C, int K, int J,
                          float* __restrict__ out, long total)
{
  long t = blockIdx.x * 256L + threadIdx.x; if (t >= total) return;
  int j = (int)(t % J); int c = (int)((t / J) % C); int b = (int)(t / ((long)C * J));
  const float* p = x + (((long)b * C + c) * K) * J + j;
  float m = p[0];
  for (int kk = 1; kk < K; ++kk) m = fmaxf(m, p[(long)kk * J]);
  out[t] = m;
}

// concat x [B,C,J] with broadcast f2 [B,F] -> out [B,C+F,J]   (addf / fp3-cat)
__global__ void k_addf(const float* __restrict__ x, const float* __restrict__ f2,
                       int C, int F, int J, float* __restrict__ out, long total)
{
  long t = blockIdx.x * 256L + threadIdx.x; if (t >= total) return;
  int j = (int)(t % J); int c = (int)((t / J) % (C + F)); int b = (int)(t / ((long)(C + F) * J));
  out[t] = (c < C) ? x[((long)b * C + c) * J + j] : f2[(long)b * F + (c - C)];
}

// concat a [B,C1,J] with b2 [B,C2,J]
__global__ void k_cat2(const float* __restrict__ a, const float* __restrict__ b2,
                       int C1, int C2, int J, float* __restrict__ out, long total)
{
  long t = blockIdx.x * 256L + threadIdx.x; if (t >= total) return;
  int j = (int)(t % J); int c = (int)((t / J) % (C1 + C2)); int b = (int)(t / ((long)(C1 + C2) * J));
  out[t] = (c < C1) ? a[((long)b * C1 + c) * J + j] : b2[((long)b * C2 + (c - C1)) * J + j];
}

// 3-NN of x1 points among x2 points + normalized inverse-distance weights
__global__ void k_knn3(const float* __restrict__ x1, const float* __restrict__ x2,
                       int Np, int S, int* __restrict__ idx3, float* __restrict__ w3,
                       long total)
{
  long t = blockIdx.x * 256L + threadIdx.x; if (t >= total) return;
  int b = (int)(t / Np), n = (int)(t % Np);
  float ax = x1[((long)b * 3 + 0) * Np + n];
  float ay = x1[((long)b * 3 + 1) * Np + n];
  float az = x1[((long)b * 3 + 2) * Np + n];
  const float* p = x2 + (long)b * 3 * S;
  float d0 = 1e30f, d1 = 1e30f, d2 = 1e30f; int i0 = 0, i1 = 0, i2 = 0;
  for (int s = 0; s < S; ++s) {
    float dx = p[s] - ax, dy = p[S + s] - ay, dz = p[2 * S + s] - az;
    float d = dx * dx + dy * dy + dz * dz;
    if (d < d0)      { d2 = d1; i2 = i1; d1 = d0; i1 = i0; d0 = d; i0 = s; }
    else if (d < d1) { d2 = d1; i2 = i1; d1 = d;  i1 = s; }
    else if (d < d2) { d2 = d;  i2 = s; }
  }
  float w0 = 1.f / (d0 + 1e-8f), w1 = 1.f / (d1 + 1e-8f), w2 = 1.f / (d2 + 1e-8f);
  float s3 = w0 + w1 + w2;
  idx3[t * 3 + 0] = i0; idx3[t * 3 + 1] = i1; idx3[t * 3 + 2] = i2;
  w3[t * 3 + 0] = w0 / s3; w3[t * 3 + 1] = w1 / s3; w3[t * 3 + 2] = w2 / s3;
}

// interpolate p2 [B,C2,S] onto Np points -> out [B,C2,Np]
__global__ void k_interp3(const float* __restrict__ p2, const int* __restrict__ idx3,
                          const float* __restrict__ w3, int C2, int Np, int S,
                          float* __restrict__ out, long total)
{
  long t = blockIdx.x * 256L + threadIdx.x; if (t >= total) return;
  int n = (int)(t % Np); int c = (int)((t / Np) % C2); int b = (int)(t / ((long)C2 * Np));
  const int*   id = idx3 + ((long)b * Np + n) * 3;
  const float* w  = w3   + ((long)b * Np + n) * 3;
  const float* p  = p2 + ((long)b * C2 + c) * S;
  out[t] = w[0] * p[id[0]] + w[1] * p[id[1]] + w[2] * p[id[2]];
}

// 16-NN (including self) over all N points, insertion sort per thread
__global__ void k_knn16(const float* __restrict__ pos, int N, int* __restrict__ idx,
                        long total)
{
  long t = blockIdx.x * 256L + threadIdx.x; if (t >= total) return;
  int b = (int)(t / N), n = (int)(t % N);
  const float* p = pos + (long)b * 3 * N;
  float ax = p[n], ay = p[N + n], az = p[2 * N + n];
  float bd[16]; int bi[16];
  for (int k = 0; k < 16; ++k) { bd[k] = 1e30f; bi[k] = 0; }
  for (int i = 0; i < N; ++i) {
    float dx = p[i] - ax, dy = p[N + i] - ay, dz = p[2 * N + i] - az;
    float d = dx * dx + dy * dy + dz * dz;
    if (d < bd[15]) {
      int k = 15;
      while (k > 0 && bd[k - 1] > d) { bd[k] = bd[k - 1]; bi[k] = bi[k - 1]; --k; }
      bd[k] = d; bi[k] = i;
    }
  }
  int* o = idx + ((long)b * N + n) * 16;
  for (int k = 0; k < 16; ++k) o[k] = bi[k];
}

// pos_rel [B,3,N,16] = pos[c,n] - pos[c, idx[n,kk]]
__global__ void k_posrel(const float* __restrict__ pos, const int* __restrict__ idx,
                         int N, float* __restrict__ out, long total)
{
  long t = blockIdx.x * 256L + threadIdx.x; if (t >= total) return;
  int kk = (int)(t & 15); long r = t >> 4;
  int n = (int)(r % N); int c = (int)((r / N) % 3); int b = (int)(r / (3L * N));
  int id = idx[((long)b * N + n) * 16 + kk];
  out[t] = pos[((long)b * 3 + c) * N + n] - pos[((long)b * 3 + c) * N + id];
}

// qk_rel + pe: out [B,C,N,16] = q[c,n] - k[c,idx[n,kk]] + pe[...]
__global__ void k_qkpe(const float* __restrict__ q, const float* __restrict__ kx,
                       const float* __restrict__ pe, const int* __restrict__ idx,
                       int C, int N, float* __restrict__ out, long total)
{
  long t = blockIdx.x * 256L + threadIdx.x; if (t >= total) return;
  int kk = (int)(t & 15); long r = t >> 4;
  int n = (int)(r % N); int c = (int)((r / N) % C); int b = (int)(r / ((long)C * N));
  int id = idx[((long)b * N + n) * 16 + kk];
  out[t] = q[((long)b * C + c) * N + n] - kx[((long)b * C + c) * N + id] + pe[t];
}

// in-place softmax over trailing 16
__global__ void k_softmax16(float* __restrict__ a, long total)
{
  long t = blockIdx.x * 256L + threadIdx.x; if (t >= total) return;
  float* p = a + t * 16;
  float m = p[0];
  for (int k = 1; k < 16; ++k) m = fmaxf(m, p[k]);
  float s = 0.f;
  for (int k = 0; k < 16; ++k) { float e = expf(p[k] - m); p[k] = e; s += e; }
  float inv = 1.f / s;
  for (int k = 0; k < 16; ++k) p[k] *= inv;
}

// agg [B,C,N] = sum_k attn[c,n,k] * (v[c,n] + pe[c,n,k])
__global__ void k_agg(const float* __restrict__ attn, const float* __restrict__ v,
                      const float* __restrict__ pe, float* __restrict__ out, long total)
{
  long t = blockIdx.x * 256L + threadIdx.x; if (t >= total) return;
  const float* at = attn + t * 16;
  const float* pp = pe + t * 16;
  float vv = v[t];
  float s = 0.f;
  for (int k = 0; k < 16; ++k) s += at[k] * (vv + pp[k]);
  out[t] = s;
}

// conv_transpose1d: fc [B,Cc,N], w (Cc,Oc,2) -> out [B,Oc,2N]
__global__ void k_convt(const float* __restrict__ fc, const float* __restrict__ w,
                        int Cc, int Oc, int N, float* __restrict__ out, long total)
{
  long t = blockIdx.x * 256L + threadIdx.x; if (t >= total) return;
  int n2 = (int)(t % (2L * N)); int o = (int)((t / (2L * N)) % Oc);
  int b = (int)(t / ((long)Oc * 2 * N));
  int n = n2 >> 1, tt = n2 & 1;
  const float* f = fc + ((long)b * Cc) * N + n;
  float s = 0.f;
  for (int c = 0; c < Cc; ++c) s += f[(long)c * N] * w[((long)c * Oc + o) * 2 + tt];
  out[t] = s;
}

// concat fc_up [B,C1,N2] with H repeated x2 [B,C2,N2/2]
__global__ void k_cat_up(const float* __restrict__ fcup, const float* __restrict__ H,
                         int C1, int C2, int N2, float* __restrict__ out, long total)
{
  long t = blockIdx.x * 256L + threadIdx.x; if (t >= total) return;
  int j = (int)(t % N2); int c = (int)((t / N2) % (C1 + C2));
  int b = (int)(t / ((long)(C1 + C2) * N2));
  out[t] = (c < C1) ? fcup[((long)b * C1 + c) * N2 + j]
                    : H[((long)b * C2 + (c - C1)) * (N2 / 2) + (j >> 1)];
}

__global__ void k_relu(const float* __restrict__ x, float* __restrict__ y, long total)
{
  long t = blockIdx.x * 256L + threadIdx.x; if (t >= total) return;
  y[t] = fmaxf(x[t], 0.f);
}

// pcd_child = repeat(coarse,2) + delta
__global__ void k_child(const float* __restrict__ coarse, const float* __restrict__ delta,
                        int N, float* __restrict__ out, long total)
{
  long t = blockIdx.x * 256L + threadIdx.x; if (t >= total) return;
  int n2 = (int)(t % (2L * N)); int c = (int)((t / (2L * N)) % 3);
  int b = (int)(t / (6L * N));
  out[t] = coarse[((long)b * 3 + c) * N + (n2 >> 1)] + delta[t];
}

// ---------------------------------------------------------------------------
// Param leaf indices (jax tree flatten: sorted dict keys, tuples in order)
// ---------------------------------------------------------------------------
#define FP1_0 0
#define FP1_1 4
#define FP2_0 8
#define FP2_1 12
#define FP3_0 16
#define FP3_1 20
#define MD1_W 24
#define MD1_B 25
#define MD2_W 26
#define MD2_B 27
#define DF_B1 28
#define DF_B2 29
#define DF_BS 30
#define DF_W1 31
#define DF_W2 32
#define DF_WS 33
#define PS1_W 34
#define PS1_B 35
#define PS2_W 36
#define PS2_B 37
#define PS_W  38
#define SA1_0 39
#define SA1_1 43
#define SA1_2 47
#define SA2_0 51
#define SA2_1 55
#define SA2_2 59
#define SA3_0 63
#define SA3_1 67
#define SA3_2 71
#define AT1   75
#define AT2_W 79
#define AT2_B 80
#define END_W 81
#define END_B 82
#define MV_B1 83
#define MV_B2 84
#define MV_BS 85
#define MV_W1 86
#define MV_W2 87
#define MV_WS 88
#define POS1  89
#define POS2_W 93
#define POS2_B 94
#define WK_W 95
#define WK_B 96
#define WQ_W 97
#define WQ_B 98
#define WV_W 99
#define WV_B 100

extern "C" void kernel_launch(void* const* d_in, const int* in_sizes, int n_in,
                              void* d_out, int out_size, void* d_ws, size_t ws_size,
                              hipStream_t stream)
{
  (void)in_sizes; (void)n_in; (void)out_size; (void)ws_size;
  const int B = 4, N = 2048, F = 512;
  const float* coarse = (const float*)d_in[0];
  const float* feat   = (const float*)d_in[1];
  const float* P[101];
  for (int i = 0; i < 101; ++i) P[i] = (const float*)d_in[2 + i];

  // bump allocator over workspace; scratch region resets per phase
  size_t off = 0;
  auto A = [&](size_t nfloats) -> float* {
    float* p = (float*)((char*)d_ws + off);
    off += (nfloats * sizeof(float) + 255) & ~(size_t)255;
    return p;
  };
  auto eg = [](long total) { return dim3((unsigned)((total + 255) / 256)); };

  auto gemm = [&](const float* W, const float* X, const float* bi, const float* g,
                  const float* be, const float* res, float* Y,
                  int O, int C, int M, int flags, long ldX, long ldY, long ldR) {
    dim3 grid((M + 63) / 64, (O + 15) / 16, B);
    gemm_wmma_kernel<<<grid, 128, 0, stream>>>(W, X, bi, g, be, res, Y,
                                               O, C, M, ldX, ldY, ldR, flags);
  };
  auto gemmc = [&](const float* W, const float* X, const float* bi, const float* g,
                   const float* be, const float* res, float* Y,
                   int O, int C, int M, int flags) {
    gemm(W, X, bi, g, be, res, Y, O, C, M, flags, M, M, M);
  };
  auto cbr = [&](int pi, const float* X, float* Y, int O, int C, int M) {
    gemmc(P[pi], X, P[pi + 1], P[pi + 2], P[pi + 3], nullptr, Y, O, C, M,
          GF_BIAS | GF_BN | GF_RELU);
  };

  // ---------------- persistent buffers ----------------
  float* l1_xyz = A((size_t)B * 3 * 512);
  float* l1p    = A((size_t)B * 640 * 512);
  float* l2_xyz = A((size_t)B * 3 * 128);
  float* l2p    = A((size_t)B * 768 * 128);
  float* l3p    = A((size_t)B * 1024);
  float* l2pn   = A((size_t)B * 768 * 128);
  float* l1pn   = A((size_t)B * 640 * 512);
  float* l0p    = A((size_t)B * 128 * 2048);
  float* value  = A((size_t)B * 128 * 2048);
  float* kbuf   = A((size_t)B * 64 * 2048);
  float* qbuf   = A((size_t)B * 64 * 2048);
  float* vbuf   = A((size_t)B * 64 * 2048);
  int*   knn16i = (int*)A((size_t)B * 2048 * 16);
  float* pe     = A((size_t)B * 64 * 2048 * 16);
  float* attn   = A((size_t)B * 64 * 2048 * 16);
  float* Hbuf   = A((size_t)B * 128 * 2048);
  float* fcup   = A((size_t)B * 128 * 4096);
  const size_t scr_base = off;

  long t;

  // ================= SA1: 2048 -> 512 points =================
  {
    off = scr_base;
    int*   fps1 = (int*)A((size_t)B * 512);
    int*   bq1  = (int*)A((size_t)B * 512 * 32);
    float* xs   = A((size_t)B * 6 * 32 * 512);
    float* s0   = A((size_t)B * 64 * 32 * 512);
    float* s1   = A((size_t)B * 64 * 32 * 512);
    float* s2   = A((size_t)B * 128 * 32 * 512);
    float* mx   = A((size_t)B * 128 * 512);
    k_fps<<<B, 256, 0, stream>>>(coarse, N, 512, fps1);
    t = (long)B * 3 * 512;
    k_gather_xyz<<<eg(t), 256, 0, stream>>>(coarse, fps1, N, 512, l1_xyz, t);
    t = (long)B * 512;
    k_ball_query<<<eg(t), 256, 0, stream>>>(coarse, l1_xyz, N, 512, 32, 0.01f, bq1, t);
    t = (long)B * 6 * 32 * 512;
    k_group_sa<<<eg(t), 256, 0, stream>>>(coarse, l1_xyz, coarse, bq1, N, 512, 32, 3, xs, t);
    const int M = 32 * 512;
    cbr(SA1_0, xs, s0, 64, 6, M);
    cbr(SA1_1, s0, s1, 64, 64, M);
    cbr(SA1_2, s1, s2, 128, 64, M);
    t = (long)B * 128 * 512;
    k_maxpool<<<eg(t), 256, 0, stream>>>(s2, 128, 32, 512, mx, t);
    t = (long)B * 640 * 512;
    k_addf<<<eg(t), 256, 0, stream>>>(mx, feat, 128, F, 512, l1p, t);
  }

  // ================= SA2: 512 -> 128 points =================
  {
    off = scr_base;
    int*   fps2 = (int*)A((size_t)B * 128);
    int*   bq2  = (int*)A((size_t)B * 128 * 32);
    float* xs   = A((size_t)B * 643 * 32 * 128);
    float* s0   = A((size_t)B * 128 * 32 * 128);
    float* s1   = A((size_t)B * 128 * 32 * 128);
    float* s2   = A((size_t)B * 256 * 32 * 128);
    float* mx   = A((size_t)B * 256 * 128);
    k_fps<<<B, 256, 0, stream>>>(l1_xyz, 512, 128, fps2);
    t = (long)B * 3 * 128;
    k_gather_xyz<<<eg(t), 256, 0, stream>>>(l1_xyz, fps2, 512, 128, l2_xyz, t);
    t = (long)B * 128;
    k_ball_query<<<eg(t), 256, 0, stream>>>(l1_xyz, l2_xyz, 512, 128, 32, 0.04f, bq2, t);
    t = (long)B * 643 * 32 * 128;
    k_group_sa<<<eg(t), 256, 0, stream>>>(l1_xyz, l2_xyz, l1p, bq2, 512, 128, 32, 640, xs, t);
    const int M = 32 * 128;
    cbr(SA2_0, xs, s0, 128, 643, M);
    cbr(SA2_1, s0, s1, 128, 128, M);
    cbr(SA2_2, s1, s2, 256, 128, M);
    t = (long)B * 256 * 128;
    k_maxpool<<<eg(t), 256, 0, stream>>>(s2, 256, 32, 128, mx, t);
    t = (long)B * 768 * 128;
    k_addf<<<eg(t), 256, 0, stream>>>(mx, feat, 256, F, 128, l2p, t);
  }

  // ================= SA3: group-all over 128 points =================
  {
    off = scr_base;
    float* xs = A((size_t)B * 771 * 128);
    float* s0 = A((size_t)B * 256 * 128);
    float* s1 = A((size_t)B * 256 * 128);
    float* s2 = A((size_t)B * 512 * 128);
    float* mx = A((size_t)B * 512);
    t = (long)B * 771 * 128;
    k_cat2<<<eg(t), 256, 0, stream>>>(l2_xyz, l2p, 3, 768, 128, xs, t);
    cbr(SA3_0, xs, s0, 256, 771, 128);
    cbr(SA3_1, s0, s1, 256, 256, 128);
    cbr(SA3_2, s1, s2, 512, 256, 128);
    t = (long)B * 512;
    k_maxpool<<<eg(t), 256, 0, stream>>>(s2, 512, 128, 1, mx, t);
    t = (long)B * 1024;
    k_addf<<<eg(t), 256, 0, stream>>>(mx, feat, 512, F, 1, l3p, t);
  }

  // ================= FP3 (S==1 broadcast) =================
  {
    off = scr_base;
    float* xc = A((size_t)B * 1792 * 128);
    float* h0 = A((size_t)B * 256 * 128);
    float* h1 = A((size_t)B * 256 * 128);
    t = (long)B * 1792 * 128;
    k_addf<<<eg(t), 256, 0, stream>>>(l2p, l3p, 768, 1024, 128, xc, t);
    cbr(FP3_0, xc, h0, 256, 1792, 128);
    cbr(FP3_1, h0, h1, 256, 256, 128);
    t = (long)B * 768 * 128;
    k_addf<<<eg(t), 256, 0, stream>>>(h1, feat, 256, F, 128, l2pn, t);
  }

  // ================= FP2 =================
  {
    off = scr_base;
    int*   i3  = (int*)A((size_t)B * 512 * 3);
    float* w3  = A((size_t)B * 512 * 3);
    float* itp = A((size_t)B * 768 * 512);
    float* xc  = A((size_t)B * 1408 * 512);
    float* h0  = A((size_t)B * 256 * 512);
    float* h1  = A((size_t)B * 128 * 512);
    t = (long)B * 512;
    k_knn3<<<eg(t), 256, 0, stream>>>(l1_xyz, l2_xyz, 512, 128, i3, w3, t);
    t = (long)B * 768 * 512;
    k_interp3<<<eg(t), 256, 0, stream>>>(l2pn, i3, w3, 768, 512, 128, itp, t);
    t = (long)B * 1408 * 512;
    k_cat2<<<eg(t), 256, 0, stream>>>(l1p, itp, 640, 768, 512, xc, t);
    cbr(FP2_0, xc, h0, 256, 1408, 512);
    cbr(FP2_1, h0, h1, 128, 256, 512);
    t = (long)B * 640 * 512;
    k_addf<<<eg(t), 256, 0, stream>>>(h1, feat, 128, F, 512, l1pn, t);
  }

  // ================= FP1 =================
  {
    off = scr_base;
    int*   i3  = (int*)A((size_t)B * 2048 * 3);
    float* w3  = A((size_t)B * 2048 * 3);
    float* itp = A((size_t)B * 640 * 2048);
    float* h0  = A((size_t)B * 128 * 2048);
    t = (long)B * 2048;
    k_knn3<<<eg(t), 256, 0, stream>>>(coarse, l1_xyz, 2048, 512, i3, w3, t);
    t = (long)B * 640 * 2048;
    k_interp3<<<eg(t), 256, 0, stream>>>(l1pn, i3, w3, 640, 2048, 512, itp, t);
    cbr(FP1_0, itp, h0, 128, 640, 2048);
    cbr(FP1_1, h0, l0p, 128, 128, 2048);
  }

  // ================= Skip transformer =================
  {
    off = scr_base;
    float* catkq = A((size_t)B * 256 * 2048);
    float* hv    = A((size_t)B * 128 * 2048);
    float* prel  = A((size_t)B * 3 * 2048 * 16);
    float* peh   = A((size_t)B * 64 * 2048 * 16);
    float* qkpe  = A((size_t)B * 64 * 2048 * 16);
    float* h256  = A((size_t)B * 256 * 512 * 16);
    float* aggb  = A((size_t)B * 64 * 2048);
    t = (long)B * 256 * 2048;
    k_cat2<<<eg(t), 256, 0, stream>>>(l0p, l0p, 128, 128, 2048, catkq, t);
    // value = mlp_res(cat): conv2(relu(conv1(x))) + sc(x)
    gemmc(P[MV_WS], catkq, P[MV_BS], 0, 0, 0, value, 128, 256, 2048, GF_BIAS);
    gemmc(P[MV_W1], catkq, P[MV_B1], 0, 0, 0, hv, 128, 256, 2048, GF_BIAS | GF_RELU);
    gemm(P[MV_W2], hv, P[MV_B2], 0, 0, value, value, 128, 128, 2048,
         GF_BIAS | GF_RES, 2048, 2048, 2048);
    gemmc(P[WK_W], l0p,   P[WK_B], 0, 0, 0, kbuf, 64, 128, 2048, GF_BIAS);
    gemmc(P[WQ_W], l0p,   P[WQ_B], 0, 0, 0, qbuf, 64, 128, 2048, GF_BIAS);
    gemmc(P[WV_W], value, P[WV_B], 0, 0, 0, vbuf, 64, 128, 2048, GF_BIAS);
    t = (long)B * 2048;
    k_knn16<<<eg(t), 256, 0, stream>>>(coarse, 2048, knn16i, t);
    t = (long)B * 3 * 2048 * 16;
    k_posrel<<<eg(t), 256, 0, stream>>>(coarse, knn16i, 2048, prel, t);
    const int M16 = 2048 * 16;
    cbr(POS1, prel, peh, 64, 3, M16);
    gemmc(P[POS2_W], peh, P[POS2_B], 0, 0, 0, pe, 64, 64, M16, GF_BIAS);
    t = (long)B * 64 * M16;
    k_qkpe<<<eg(t), 256, 0, stream>>>(qbuf, kbuf, pe, knn16i, 64, 2048, qkpe, t);
    // attention MLP 64 -> 256 -> 64, chunked over points to cap workspace
    for (int n0 = 0; n0 < 2048; n0 += 512) {
      const int Mc = 512 * 16;
      gemm(P[AT1], qkpe + (long)n0 * 16, P[AT1 + 1], P[AT1 + 2], P[AT1 + 3], 0,
           h256, 256, 64, Mc, GF_BIAS | GF_BN | GF_RELU, (long)M16, (long)Mc, 0);
      gemm(P[AT2_W], h256, P[AT2_B], 0, 0, 0, attn + (long)n0 * 16,
           64, 256, Mc, GF_BIAS, (long)Mc, (long)M16, 0);
    }
    t = (long)B * 64 * 2048;
    k_softmax16<<<eg(t), 256, 0, stream>>>(attn, t);
    k_agg<<<eg(t), 256, 0, stream>>>(attn, vbuf, pe, aggb, t);
    gemm(P[END_W], aggb, P[END_B], 0, 0, value, Hbuf, 128, 64, 2048,
         GF_BIAS | GF_RES, 2048, 2048, 2048);
  }

  // ================= Upsample + displacement =================
  {
    off = scr_base;
    float* h64   = A((size_t)B * 64 * 2048);
    float* fc32  = A((size_t)B * 32 * 2048);
    float* catdf = A((size_t)B * 256 * 4096);
    float* scb   = A((size_t)B * 128 * 4096);
    float* hdf   = A((size_t)B * 128 * 4096);
    float* rbuf  = A((size_t)B * 128 * 4096);
    float* m64   = A((size_t)B * 64 * 4096);
    float* delta = A((size_t)B * 3 * 4096);
    float* out   = (float*)d_out;
    float* Kcur  = out + (size_t)B * 3 * 4096;
    gemmc(P[PS1_W], Hbuf, P[PS1_B], 0, 0, 0, h64, 64, 128, 2048, GF_BIAS | GF_RELU);
    gemmc(P[PS2_W], h64, P[PS2_B], 0, 0, 0, fc32, 32, 64, 2048, GF_BIAS);
    t = (long)B * 128 * 4096;
    k_convt<<<eg(t), 256, 0, stream>>>(fc32, P[PS_W], 32, 128, 2048, fcup, t);
    t = (long)B * 256 * 4096;
    k_cat_up<<<eg(t), 256, 0, stream>>>(fcup, Hbuf, 128, 128, 4096, catdf, t);
    // K_curr = mlp_res(cat) written directly into d_out
    gemmc(P[DF_WS], catdf, P[DF_BS], 0, 0, 0, scb, 128, 256, 4096, GF_BIAS);
    gemmc(P[DF_W1], catdf, P[DF_B1], 0, 0, 0, hdf, 128, 256, 4096, GF_BIAS | GF_RELU);
    gemm(P[DF_W2], hdf, P[DF_B2], 0, 0, scb, Kcur, 128, 128, 4096,
         GF_BIAS | GF_RES, 4096, 4096, 4096);
    t = (long)B * 128 * 4096;
    k_relu<<<eg(t), 256, 0, stream>>>(Kcur, rbuf, t);
    gemmc(P[MD1_W], rbuf, P[MD1_B], 0, 0, 0, m64, 64, 128, 4096, GF_BIAS | GF_RELU);
    gemmc(P[MD2_W], m64, P[MD2_B], 0, 0, 0, delta, 3, 64, 4096, GF_BIAS | GF_TANH);
    t = (long)B * 3 * 4096;
    k_child<<<eg(t), 256, 0, stream>>>(coarse, delta, 2048, out, t);
  }
}
